// ATSSPostProcessor_66434554135141
// MI455X (gfx1250) — compile-verified, matrix operand order checked by reading, not run on hardware
//
#include <hip/hip_runtime.h>
#include <math.h>

// ---------------- constants mirroring the reference ----------------
#define NIMG   8
#define HWLOC  15200      // 100*152
#define CNUM   80
#define LTOK   256
#define PRE_THRESH   0.05f
#define PRE_TOPN     1000
#define NMS_THRESH   0.6f
#define POST_TOPN    100
#define IMG_W_F      1216.0f
#define IMG_H_F      800.0f
#define CAP          4096   // compaction capacity per image (>= top-1000 + bin slack)
#define NBIN         1024

typedef _Float16 half_t;
typedef __attribute__((ext_vector_type(16))) _Float16 v16h;
typedef __attribute__((ext_vector_type(8)))  float    v8f;

// Fast sigmoid: v_exp_f32 + v_add + v_rcp_f32 (result feeds f16, ~1ulp rcp is plenty)
__device__ __forceinline__ float sigmoidf_(float x) {
    return __builtin_amdgcn_rcpf(1.0f + __expf(-x));
}

// K index of element pair j of the 16x32 f16 A/B operand, per CDNA5 ISA 7.12.2:
// VGPR j (j<4): lanes0-15 K=2j,2j+1 ; lanes16-31 K=8+2j,8+2j+1 ; VGPR 4-7: +16.
__device__ __forceinline__ int wmma_k_pair_base(int j, int half_sel) {
    return (j < 4 ? 0 : 16) + half_sel * 8 + 2 * (j & 3);
}

// ---------------- K0: zero histograms + counters ----------------
__global__ void atss_init_kernel(unsigned* __restrict__ hist,
                                 unsigned* __restrict__ cnt) {
    int t = blockIdx.x * 256 + threadIdx.x;
    if (t < NIMG * NBIN) hist[t] = 0u;
    if (t < NIMG)        cnt[t]  = 0u;
}

// ---------------- K0b: pre-swizzle pos_map into per-lane WMMA-B layout ----------
// bws[((ct*8+kk)*32 + lane)*16 + e] = f16(pos[ct*16 + (lane&15)][kk*32 + K(e,lane>>4)])
__global__ void atss_bprep_kernel(const float* __restrict__ pos,
                                  half_t* __restrict__ bws) {
    int t = blockIdx.x * 256 + threadIdx.x;
    if (t >= 5 * 8 * 32) return;
    int lane = t & 31;
    int kk   = (t >> 5) & 7;
    int ct   = t >> 8;
    int half_sel = lane >> 4, ml = lane & 15;
    const float* prow = pos + (size_t)(ct * 16 + ml) * LTOK;
    half_t* o = bws + (size_t)t * 16;
    #pragma unroll
    for (int j = 0; j < 8; ++j) {
        int k = kk * 32 + wmma_k_pair_base(j, half_sel);
        o[2 * j]     = (half_t)prow[k];
        o[2 * j + 1] = (half_t)prow[k + 1];
    }
}

// ---------------- K1: WMMA score GEMM + fused epilogue ----------------
// scores[n,hw,c] = sum_l sigmoid(dp[n,hw,l]) * pos[c,l]
// masked[n,hw,c] = (scores>0.05) ? scores*sigmoid(ctr[n,hw]) : -1
// + per-image 1024-bin histogram of masked>=0 values.
__global__ __launch_bounds__(256)
void atss_score_gemm_kernel(const float* __restrict__ dp,     // [N,HW,L]
                            const v16h*  __restrict__ bmat,   // [5*8*32] swizzled B
                            const float* __restrict__ ctr,    // [N,HW]
                            float* __restrict__ masked,       // [N,HW,C]
                            unsigned* __restrict__ hist) {    // [N,1024]
    __shared__ unsigned lhist[NBIN];
    for (int b = threadIdx.x; b < NBIN; b += 256) lhist[b] = 0u;
    __syncthreads();

    const int n    = blockIdx.y;
    const int wave = threadIdx.x >> 5;          // 8 waves / block (wave32)
    const int lane = threadIdx.x & 31;
    const int rt   = blockIdx.x * 8 + wave;     // row tile id, 950 tiles
    if (rt < (HWLOC / 16)) {
        const int half_sel = lane >> 4;         // lane half select (0/1)
        const int ml       = lane & 15;
        const int rowA     = rt * 16 + ml;      // A-matrix row held by this lane
        const float* dprow = dp + ((size_t)n * HWLOC + rowA) * LTOK;

        // Load + sigmoid + cvt A for all 8 K-steps (held in 64 VGPRs).
        v16h A[8];
        #pragma unroll
        for (int kk = 0; kk < 8; ++kk) {
            const int kbase = kk * 32;
            #pragma unroll
            for (int j = 0; j < 8; ++j) {
                int k = kbase + wmma_k_pair_base(j, half_sel);
                float f0 = sigmoidf_(dprow[k]);
                float f1 = sigmoidf_(dprow[k + 1]);
                A[kk][2 * j]     = (half_t)f0;
                A[kk][2 * j + 1] = (half_t)f1;
            }
        }

        // Centerness for the 8 rows this lane's accumulators map to.
        float ctrv[8];
        #pragma unroll
        for (int r = 0; r < 8; ++r) {
            int row = rt * 16 + r + half_sel * 8;
            ctrv[r] = sigmoidf_(ctr[(size_t)n * HWLOC + row]);
        }

        #pragma unroll
        for (int ct = 0; ct < 5; ++ct) {        // 5 column tiles of 16 classes
            v8f acc = {};
            #pragma unroll
            for (int kk = 0; kk < 8; ++kk) {
                v16h B = bmat[(ct * 8 + kk) * 32 + lane];   // 2x b128, L2-resident
                acc = __builtin_amdgcn_wmma_f32_16x16x32_f16(
                    false, A[kk], false, B, (short)0, acc, false, false);
            }
            // Epilogue: VGPR r -> (M = r + 8*half_sel, Nc = lane&15)
            #pragma unroll
            for (int r = 0; r < 8; ++r) {
                int row = rt * 16 + r + half_sel * 8;
                int cls = ct * 16 + ml;
                float s = acc[r];
                float v = (s > PRE_THRESH) ? s * ctrv[r] : -1.0f;
                masked[((size_t)n * HWLOC + row) * CNUM + cls] = v;
                if (v >= 0.0f) {
                    int bin = (int)(v * (float)NBIN);
                    if (bin > NBIN - 1) bin = NBIN - 1;
                    atomicAdd(&lhist[bin], 1u);
                }
            }
        }
    }
    __syncthreads();
    for (int b = threadIdx.x; b < NBIN; b += 256) {
        unsigned c = lhist[b];
        if (c) atomicAdd(&hist[(size_t)n * NBIN + b], c);
    }
}

// ---------------- K2: per-image histogram threshold ----------------
__global__ void atss_thresh_kernel(const unsigned* __restrict__ hist,
                                   unsigned* __restrict__ thresh) {
    int n = threadIdx.x;
    if (n >= NIMG) return;
    unsigned cum = 0, t = 0;
    for (int b = NBIN - 1; b >= 0; --b) {
        cum += hist[n * NBIN + b];
        if (cum >= (unsigned)PRE_TOPN) { t = (unsigned)b; break; }
    }
    thresh[n] = t;
}

// ---------------- K3: compact survivors (score>=bin threshold) ----------------
__global__ void atss_compact_kernel(const float* __restrict__ masked,
                                    const unsigned* __restrict__ thresh,
                                    unsigned* __restrict__ cnt,
                                    unsigned long long* __restrict__ keys) {
    size_t t = (size_t)blockIdx.x * 256 + threadIdx.x;
    const size_t PER = (size_t)HWLOC * CNUM;
    if (t >= (size_t)NIMG * PER) return;
    int n = (int)(t / PER);
    unsigned local = (unsigned)(t % PER);   // flat idx = loc*C + cls
    float v = masked[t];
    if (v >= 0.0f && v * (float)NBIN >= (float)thresh[n]) {
        unsigned pos = atomicAdd(&cnt[n], 1u);
        if (pos < CAP)
            keys[(size_t)n * CAP + pos] =
                ((unsigned long long)__float_as_uint(v) << 32) | (unsigned long long)local;
    }
}

// ---------------- K4: per-image bitonic sort (desc) + top-1000 ----------------
__global__ __launch_bounds__(1024)
void atss_sort_kernel(const unsigned long long* __restrict__ keys,
                      const unsigned* __restrict__ cnt,
                      unsigned long long* __restrict__ topk) {
    __shared__ unsigned long long s[CAP];   // 32 KB LDS
    int n = blockIdx.x;
    unsigned M = cnt[n]; if (M > CAP) M = CAP;
    for (int i = threadIdx.x; i < CAP; i += 1024)
        s[i] = (i < (int)M) ? keys[(size_t)n * CAP + i] : 0ull;
    __syncthreads();
    for (unsigned k = 2; k <= CAP; k <<= 1) {
        for (unsigned j = k >> 1; j > 0; j >>= 1) {
            for (unsigned i = threadIdx.x; i < CAP; i += 1024) {
                unsigned ixj = i ^ j;
                if (ixj > i) {
                    bool desc = ((i & k) == 0);
                    unsigned long long a = s[i], b = s[ixj];
                    if (desc ? (a < b) : (a > b)) { s[i] = b; s[ixj] = a; }
                }
            }
            __syncthreads();
        }
    }
    for (int i = threadIdx.x; i < PRE_TOPN; i += 1024)
        topk[(size_t)n * PRE_TOPN + i] = s[i];
}

// ---------------- K5: box decode ----------------
__global__ void atss_decode_kernel(const unsigned long long* __restrict__ topk,
                                   const float* __restrict__ anchors,   // [HW,4]
                                   const float* __restrict__ reg,       // [N,4,H,W]
                                   float* __restrict__ dec) {           // [N,1000,8]
    int t = blockIdx.x * 256 + threadIdx.x;
    if (t >= NIMG * PRE_TOPN) return;
    int n = t / PRE_TOPN;
    unsigned long long key = topk[t];
    float v = __uint_as_float((unsigned)(key >> 32));
    unsigned idx = (unsigned)(key & 0xffffffffu);
    float* o = dec + (size_t)t * 8;
    if (key == 0ull || !(v > 0.0f)) {
        #pragma unroll
        for (int q = 0; q < 8; ++q) o[q] = 0.0f;
        return;
    }
    const float MAXD = logf(1000.0f / 16.0f);
    int loc = (int)(idx / CNUM);
    int cls = (int)(idx % CNUM);
    const float* a = anchors + (size_t)loc * 4;
    float aw = a[2] - a[0], ah = a[3] - a[1];
    float acx = a[0] + 0.5f * aw, acy = a[1] + 0.5f * ah;
    float r0 = reg[((size_t)n * 4 + 0) * HWLOC + loc];
    float r1 = reg[((size_t)n * 4 + 1) * HWLOC + loc];
    float r2 = reg[((size_t)n * 4 + 2) * HWLOC + loc];
    float r3 = reg[((size_t)n * 4 + 3) * HWLOC + loc];
    float dx = r0 * 0.1f, dy = r1 * 0.1f;                 // /10
    float dw = fminf(r2 * 0.2f, MAXD);                    // /5, clamp
    float dh = fminf(r3 * 0.2f, MAXD);
    float pcx = dx * aw + acx, pcy = dy * ah + acy;
    float pw = __expf(dw) * aw, ph = __expf(dh) * ah;
    float x1 = fminf(fmaxf(pcx - 0.5f * pw, 0.0f), IMG_W_F);
    float y1 = fminf(fmaxf(pcy - 0.5f * ph, 0.0f), IMG_H_F);
    float x2 = fminf(fmaxf(pcx + 0.5f * pw, 0.0f), IMG_W_F);
    float y2 = fminf(fmaxf(pcy + 0.5f * ph, 0.0f), IMG_H_F);
    int valid = ((x2 - x1) > 0.0f) && ((y2 - y1) > 0.0f);
    o[0] = x1; o[1] = y1; o[2] = x2; o[3] = y2;
    o[4] = sqrtf(v);
    o[5] = (float)(cls + 1);
    o[6] = valid ? 1.0f : 0.0f;
    o[7] = 0.0f;
}

// ---------------- K6: class-aware greedy NMS + top-100 writeback ----------------
__global__ __launch_bounds__(1024)
void atss_nms_kernel(const float* __restrict__ dec,   // [N,1000,8] sorted desc
                     float* __restrict__ out) {
    __shared__ float bx1[PRE_TOPN], by1[PRE_TOPN], bx2[PRE_TOPN], by2[PRE_TOPN];
    __shared__ float bs[PRE_TOPN], barea[PRE_TOPN];
    __shared__ int   blab[PRE_TOPN], bkeep[PRE_TOPN];
    int n = blockIdx.x, tid = threadIdx.x;
    if (tid < PRE_TOPN) {
        const float* o = dec + ((size_t)n * PRE_TOPN + tid) * 8;
        bx1[tid] = o[0]; by1[tid] = o[1]; bx2[tid] = o[2]; by2[tid] = o[3];
        bs[tid]  = o[4]; blab[tid] = (int)o[5];
        bkeep[tid] = (o[6] > 0.0f) ? 1 : 0;
        barea[tid] = (o[2] - o[0]) * (o[3] - o[1]);
    }
    __syncthreads();
    for (int i = 0; i < PRE_TOPN - 1; ++i) {
        if (bkeep[i]) {   // uniform: all threads read same LDS value
            int j = tid;
            if (j > i && j < PRE_TOPN && bkeep[j] && blab[j] == blab[i]) {
                float ix1 = fmaxf(bx1[i], bx1[j]);
                float iy1 = fmaxf(by1[i], by1[j]);
                float ix2 = fminf(bx2[i], bx2[j]);
                float iy2 = fminf(by2[i], by2[j]);
                float inter = fmaxf(ix2 - ix1, 0.0f) * fmaxf(iy2 - iy1, 0.0f);
                float iou = inter / (barea[i] + barea[j] - inter + 1e-6f);
                if (iou > NMS_THRESH) bkeep[j] = 0;
            }
        }
        __syncthreads();
    }
    if (tid == 0) {
        float* ob = out;                                       // [N,100,4]
        float* os = out + NIMG * POST_TOPN * 4;                // [N,100]
        float* ol = out + NIMG * POST_TOPN * 4 + NIMG * POST_TOPN;
        int slot = 0;
        for (int i = 0; i < PRE_TOPN && slot < POST_TOPN; ++i) {
            if (bkeep[i] && bs[i] > 0.0f) {
                ob[((size_t)n * POST_TOPN + slot) * 4 + 0] = bx1[i];
                ob[((size_t)n * POST_TOPN + slot) * 4 + 1] = by1[i];
                ob[((size_t)n * POST_TOPN + slot) * 4 + 2] = bx2[i];
                ob[((size_t)n * POST_TOPN + slot) * 4 + 3] = by2[i];
                os[n * POST_TOPN + slot] = bs[i];
                ol[n * POST_TOPN + slot] = (float)blab[i];
                ++slot;
            }
        }
        for (; slot < POST_TOPN; ++slot) {
            ob[((size_t)n * POST_TOPN + slot) * 4 + 0] = 0.0f;
            ob[((size_t)n * POST_TOPN + slot) * 4 + 1] = 0.0f;
            ob[((size_t)n * POST_TOPN + slot) * 4 + 2] = 0.0f;
            ob[((size_t)n * POST_TOPN + slot) * 4 + 3] = 0.0f;
            os[n * POST_TOPN + slot] = 0.0f;
            ol[n * POST_TOPN + slot] = 0.0f;
        }
    }
}

// ---------------- launch ----------------
extern "C" void kernel_launch(void* const* d_in, const int* in_sizes, int n_in,
                              void* d_out, int out_size, void* d_ws, size_t ws_size,
                              hipStream_t stream) {
    const float* box_reg = (const float*)d_in[0];   // [N,4,H,W]
    const float* ctr     = (const float*)d_in[1];   // [N,1,H,W]
    // d_in[2] = box_cls: unused by the reference
    const float* dp      = (const float*)d_in[3];   // [N,HW,L]
    const float* anchors = (const float*)d_in[4];   // [HW,4]
    const float* posmap  = (const float*)d_in[5];   // [C,L]
    float* out = (float*)d_out;

    // workspace layout (bytes, 256-aligned)
    char* ws = (char*)d_ws;
    float*              masked = (float*)(ws + 0);                      // 38,912,000
    unsigned*           hist   = (unsigned*)(ws + 38912000);            // 32,768
    unsigned*           thresh = (unsigned*)(ws + 38944768);            // 32 (+pad)
    unsigned*           cnt    = (unsigned*)(ws + 38945024);            // 32 (+pad)
    unsigned long long* keys   = (unsigned long long*)(ws + 38945280);  // 262,144
    unsigned long long* topk   = (unsigned long long*)(ws + 39207424);  // 64,000
    float*              dec    = (float*)(ws + 39271424);               // 256,000
    half_t*             bws    = (half_t*)(ws + 39527424);              // 40,960

    atss_init_kernel<<<33, 256, 0, stream>>>(hist, cnt);
    atss_bprep_kernel<<<5, 256, 0, stream>>>(posmap, bws);

    dim3 g1((HWLOC / 16 + 7) / 8, NIMG);   // 119 x 8 blocks, 8 waves each
    atss_score_gemm_kernel<<<g1, 256, 0, stream>>>(dp, (const v16h*)bws, ctr,
                                                   masked, hist);

    atss_thresh_kernel<<<1, NIMG, 0, stream>>>(hist, thresh);

    size_t total = (size_t)NIMG * HWLOC * CNUM;
    atss_compact_kernel<<<(unsigned)((total + 255) / 256), 256, 0, stream>>>(
        masked, thresh, cnt, keys);

    atss_sort_kernel<<<NIMG, 1024, 0, stream>>>(keys, cnt, topk);

    atss_decode_kernel<<<(NIMG * PRE_TOPN + 255) / 256, 256, 0, stream>>>(
        topk, anchors, box_reg, dec);

    atss_nms_kernel<<<NIMG, 1024, 0, stream>>>(dec, out);
}
